// NTXentLoss_6433861010051
// MI455X (gfx1250) — matrix operand order repaired and checked by reading, not compiled, not run on hardware
//
#include <hip/hip_runtime.h>
#include <stdint.h>

typedef __attribute__((ext_vector_type(16))) __bf16 v16bf;
typedef __attribute__((ext_vector_type(8)))  __bf16 v8bf;
typedef __attribute__((ext_vector_type(8)))  float  v8f;
typedef int v4i __attribute__((__vector_size__(16)));  // matches builtin proto

#define NROWS    8192   // 2N
#define NHALF    4096   // N
#define DDIM     256
#define TEMPINV  2.0f   // 1 / 0.5
#define NSPLIT   4      // column splits (grid.y)
#define COLSPAN  (NROWS / NSPLIT)    // 2048 columns per block
#define TILES    (COLSPAN / 16)      // 128 column tiles per block

// LDS staging geometry: B tile = 16 rows x 512B (hi) + 16 x 512B (lo).
// Row stride padded 512 -> 544 B (== 8 dwords mod 64 banks) to avoid the
// 16-lane same-bank conflict on the 32B ds_load_b128 pairs.
#define LROW     544
#define LPREC    (16 * LROW)         // 8704 B per precision
#define LBUF     (2 * LPREC)         // 17408 B per double-buffer slot

#define AS1 __attribute__((address_space(1)))
#define AS3 __attribute__((address_space(3)))

#if __has_builtin(__builtin_amdgcn_global_load_async_to_lds_b128)
#define USE_ASYNC_LDS 1
#else
#define USE_ASYNC_LDS 0
#endif

__device__ __forceinline__ void wait_async0() {
#if __has_builtin(__builtin_amdgcn_s_wait_asynccnt)
  __builtin_amdgcn_s_wait_asynccnt(0);
#else
  asm volatile("s_wait_asynccnt 0" ::: "memory");
#endif
}

// ---------------------------------------------------------------------------
// Kernel 1: L2-normalize each row, split into bf16 hi + bf16 lo residual.
// One wave per row; 8 waves per block.
// ---------------------------------------------------------------------------
__global__ __launch_bounds__(256) void nrm_split_kernel(
    const float* __restrict__ z1, const float* __restrict__ z2,
    __bf16* __restrict__ zhi, __bf16* __restrict__ zlo) {
  const int lane = threadIdx.x & 31;
  const int wave = threadIdx.x >> 5;
  const int row  = blockIdx.x * 8 + wave;
  const float* src = (row < NHALF) ? (z1 + (size_t)row * DDIM)
                                   : (z2 + (size_t)(row - NHALF) * DDIM);
  float v[8];
  float ss = 0.f;
#pragma unroll
  for (int k = 0; k < 8; ++k) {
    v[k] = src[lane + 32 * k];
    ss += v[k] * v[k];
  }
#pragma unroll
  for (int m = 16; m >= 1; m >>= 1) ss += __shfl_xor(ss, m, 32);
  const float rn = 1.0f / sqrtf(ss);
#pragma unroll
  for (int k = 0; k < 8; ++k) {
    float x  = v[k] * rn;
    __bf16 h = (__bf16)x;
    float  r = x - (float)h;
    zhi[(size_t)row * DDIM + lane + 32 * k] = h;
    zlo[(size_t)row * DDIM + lane + 32 * k] = (__bf16)r;
  }
}

// ---------------------------------------------------------------------------
// Kernel 2: fused (z z^T)/T + masked online sum-exp + pos gather over a
// 128-row x 2048-col stripe. A (hi+lo, K=256) resident in VGPRs per wave.
// B tiles double-buffered in LDS via async global->LDS copies.
// ---------------------------------------------------------------------------
__global__ __launch_bounds__(256) void ntxent_fused_kernel(
    const __bf16* __restrict__ zhi, const __bf16* __restrict__ zlo,
    float* __restrict__ rowsum, float* __restrict__ rowpos) {
  const int lane = threadIdx.x & 31;
  const int wave = threadIdx.x >> 5;
  const int m    = lane & 15;
  const int half = lane >> 4;
  const int i0   = blockIdx.x * 128 + wave * 16;
  const int colbase = blockIdx.y * COLSPAN;

  __shared__ __align__(64) unsigned char ldsB[2 * LBUF];
  char* ldsbase = (char*)&ldsB[0];

  // ---- per-thread fill assignment: 4 x 16B chunks (64B of one row) ----
  const int ft    = threadIdx.x;
  const int fprec = ft >> 7;             // 0 = hi, 1 = lo
  const int frow  = (ft & 127) >> 3;     // 0..15
  const int fc16  = (ft & 7) * 4;        // 16B-chunk column base
  const char* fsrc0 =
      (const char*)(fprec ? zlo : zhi) +
      (size_t)(colbase + frow) * (DDIM * 2) + (size_t)fc16 * 16;
  const uint32_t fdst = (uint32_t)(fprec * LPREC + frow * LROW + fc16 * 16);
#if !USE_ASYNC_LDS
  uint4 stage[4];
#endif

  // ---- resident A fragments (this wave's 16 rows, full K, hi & lo) ----
  v16bf ah[8], al[8];
  {
    const __bf16* arh = zhi + (size_t)(i0 + m) * DDIM;
    const __bf16* arl = zlo + (size_t)(i0 + m) * DDIM;
#pragma unroll
    for (int c = 0; c < 8; ++c) {
      const int k0 = c * 32 + half * 16;
      v8bf h0 = *(const v8bf*)(arh + k0);
      v8bf h1 = *(const v8bf*)(arh + k0 + 8);
      ah[c] = __builtin_shufflevector(h0, h1, 0,1,2,3,4,5,6,7,8,9,10,11,12,13,14,15);
      v8bf l0 = *(const v8bf*)(arl + k0);
      v8bf l1 = *(const v8bf*)(arl + k0 + 8);
      al[c] = __builtin_shufflevector(l0, l1, 0,1,2,3,4,5,6,7,8,9,10,11,12,13,14,15);
    }
  }

  const int ibase = i0 + half * 8;  // C-layout row base for this lane
  float sumExp[8], posV[8];
#pragma unroll
  for (int r = 0; r < 8; ++r) { sumExp[r] = 0.f; posV[r] = 0.f; }

  // ---- prologue: fill buffer 0 with tile 0 ----
  {
    const char* s = fsrc0;
    char* d = ldsbase + fdst;
#if USE_ASYNC_LDS
#pragma unroll
    for (int q = 0; q < 4; ++q)
      __builtin_amdgcn_global_load_async_to_lds_b128(
          (AS1 v4i*)(s + q * 16), (AS3 v4i*)(d + q * 16), 0, 0);
    wait_async0();
#else
#pragma unroll
    for (int q = 0; q < 4; ++q) stage[q] = *(const uint4*)(s + q * 16);
#pragma unroll
    for (int q = 0; q < 4; ++q) *(uint4*)(d + q * 16) = stage[q];
#endif
    __syncthreads();
  }

  int cur = 0;
  for (int t = 0; t < TILES; ++t) {
    // ---- kick off fill of the other buffer for tile t+1 ----
    if (t + 1 < TILES) {
      const char* s = fsrc0 + (size_t)(t + 1) * (16 * DDIM * 2);
      char* d = ldsbase + (uint32_t)((cur ^ 1) * LBUF) + fdst;
#if USE_ASYNC_LDS
#pragma unroll
      for (int q = 0; q < 4; ++q)
        __builtin_amdgcn_global_load_async_to_lds_b128(
            (AS1 v4i*)(s + q * 16), (AS3 v4i*)(d + q * 16), 0, 0);
#else
#pragma unroll
      for (int q = 0; q < 4; ++q) stage[q] = *(const uint4*)(s + q * 16);
#endif
    }

    // ---- compute tile t from buffer `cur` ----
    {
      const char* bhrow = ldsbase + cur * LBUF + m * LROW + half * 32;
      const char* blrow = bhrow + LPREC;
      v8f acc0 = {};  // hi*hi
      v8f acc1 = {};  // hi*lo
      v8f acc2 = {};  // lo*hi
#pragma unroll
      for (int c = 0; c < 8; ++c) {
        v16bf bh = *(const v16bf*)(bhrow + c * 64);
        v16bf bl = *(const v16bf*)(blrow + c * 64);
        acc0 = __builtin_amdgcn_wmma_f32_16x16x32_bf16(false, ah[c], false, bh,
                                                       (short)0, acc0, false, false);
        acc1 = __builtin_amdgcn_wmma_f32_16x16x32_bf16(false, ah[c], false, bl,
                                                       (short)0, acc1, false, false);
        acc2 = __builtin_amdgcn_wmma_f32_16x16x32_bf16(false, al[c], false, bh,
                                                       (short)0, acc2, false, false);
      }
      const int j = colbase + t * 16 + m;  // this lane's global column
#pragma unroll
      for (int r = 0; r < 8; ++r) {
        const int i = ibase + r;
        const float s = TEMPINV * (acc0[r] + acc1[r] + acc2[r]);
        const float e = __expf(s);              // s in [-2,2]: no max-shift needed
        sumExp[r] += (j == i) ? 0.f : e;        // diagonal mask
        posV[r]   += (j == (i ^ NHALF)) ? s : 0.f;  // positive pair = i XOR N
      }
    }

    // ---- commit next fill, sync ----
    if (t + 1 < TILES) {
#if USE_ASYNC_LDS
      wait_async0();
#else
      char* d = ldsbase + (uint32_t)((cur ^ 1) * LBUF) + fdst;
#pragma unroll
      for (int q = 0; q < 4; ++q) *(uint4*)(d + q * 16) = stage[q];
#endif
      __syncthreads();
    }
    cur ^= 1;
  }

  // ---- reduce across the 16 lanes sharing each row ----
#pragma unroll
  for (int r = 0; r < 8; ++r) {
#pragma unroll
    for (int mm = 1; mm < 16; mm <<= 1) {
      sumExp[r] += __shfl_xor(sumExp[r], mm, 32);
      posV[r]   += __shfl_xor(posV[r],   mm, 32);
    }
  }
  // lanes 0 and 16 hold the 16 row partials -> workspace (per column-split)
  if (m == 0) {
    float* rs = rowsum + (size_t)blockIdx.y * NROWS;
    float* rp = rowpos + (size_t)blockIdx.y * NROWS;
#pragma unroll
    for (int r = 0; r < 8; ++r) {
      rs[ibase + r] = sumExp[r];
      rp[ibase + r] = posV[r];
    }
  }
}

// ---------------------------------------------------------------------------
// Kernel 3: combine column-split partials: loss = mean(log(sum) - pos).
// Single block, deterministic tree reduction.
// ---------------------------------------------------------------------------
__global__ __launch_bounds__(256) void ntxent_loss_kernel(
    const float* __restrict__ rowsum, const float* __restrict__ rowpos,
    float* __restrict__ out) {
  const int lane = threadIdx.x & 31;
  const int wave = threadIdx.x >> 5;
  float local = 0.f;
#pragma unroll
  for (int k = 0; k < NROWS / 256; ++k) {
    const int row = threadIdx.x + 256 * k;
    float s = 0.f, p = 0.f;
#pragma unroll
    for (int b = 0; b < NSPLIT; ++b) {
      s += rowsum[(size_t)b * NROWS + row];
      p += rowpos[(size_t)b * NROWS + row];
    }
    local += logf(s) - p;
  }
#pragma unroll
  for (int mm = 16; mm >= 1; mm >>= 1) local += __shfl_xor(local, mm, 32);
  __shared__ float red[8];
  if (lane == 0) red[wave] = local;
  __syncthreads();
  if (threadIdx.x == 0) {
    float tot = 0.f;
#pragma unroll
    for (int w = 0; w < 8; ++w) tot += red[w];
    out[0] = tot * (1.0f / (float)NROWS);
  }
}

// ---------------------------------------------------------------------------
extern "C" void kernel_launch(void* const* d_in, const int* in_sizes, int n_in,
                              void* d_out, int out_size, void* d_ws, size_t ws_size,
                              hipStream_t stream) {
  (void)in_sizes; (void)n_in; (void)out_size; (void)ws_size;
  const float* z1 = (const float*)d_in[0];
  const float* z2 = (const float*)d_in[1];

  // ws: zhi [4MB] | zlo [4MB] | rowsum [4*8192 f32] | rowpos [4*8192 f32]
  __bf16* zhi = (__bf16*)d_ws;
  __bf16* zlo = zhi + (size_t)NROWS * DDIM;
  float*  rowsum = (float*)(zlo + (size_t)NROWS * DDIM);
  float*  rowpos = rowsum + (size_t)NSPLIT * NROWS;

  nrm_split_kernel<<<NROWS / 8, 256, 0, stream>>>(z1, z2, zhi, zlo);
  ntxent_fused_kernel<<<dim3(NROWS / 128, NSPLIT), 256, 0, stream>>>(
      zhi, zlo, rowsum, rowpos);
  ntxent_loss_kernel<<<1, 256, 0, stream>>>(rowsum, rowpos, (float*)d_out);
}
